// TorchKMeansVectorizedCluster_60413009985968
// MI455X (gfx1250) — compile-verified, matrix operand order checked by reading, not run on hardware
//
#include <hip/hip_runtime.h>

// ---------------------------------------------------------------------------
// K-means on MI455X (gfx1250, wave32).
// Distance GEMM X(262144x128) @ C^T(128x64) done with v_wmma_f32_16x16x32_bf16.
// X (128 MB) is L2-resident (192 MB L2), so bf16 WMMA keeps us compute-fed.
// Convergence handled with a device-side flag so the whole 50-iteration loop
// is graph-capturable with no host synchronization.
// ---------------------------------------------------------------------------

#define N_POINTS      262144
#define DIM           128
#define NCLUST        64
#define MAX_ITERS     50
#define KM_TOL        1e-4f
#define PTS_PER_BLOCK 128     // 8 waves * 16 points
#define BLK           256

typedef __attribute__((ext_vector_type(16))) __bf16 v16bf;
typedef __attribute__((ext_vector_type(8)))  __bf16 v8bf;
typedef __attribute__((ext_vector_type(8)))  float  v8f;

// ---------------------------------------------------------------------------
// init: C = prev_centers, Cbf16 = bf16(C), cc[k] = ||C_k||^2, zero sums/counts
// ---------------------------------------------------------------------------
__global__ __launch_bounds__(256) void kmeans_init(
    const float* __restrict__ prev, float* __restrict__ C,
    float* __restrict__ sums, float* __restrict__ counts,
    float* __restrict__ cc, __bf16* __restrict__ Cbf, int* __restrict__ done) {
  __shared__ float scc[NCLUST];
  const int tid = threadIdx.x;
  if (tid < NCLUST) scc[tid] = 0.0f;
  __syncthreads();
  for (int e = tid; e < NCLUST * DIM; e += BLK) {
    float v = prev[e];
    C[e] = v;
    Cbf[e] = (__bf16)v;
    sums[e] = 0.0f;
    atomicAdd(&scc[e >> 7], v * v);
  }
  __syncthreads();
  if (tid < NCLUST) { cc[tid] = scc[tid]; counts[tid] = 0.0f; }
  if (tid == 0) *done = 0;
}

// ---------------------------------------------------------------------------
// assign: labels = argmin_k( cc[k] - 2 * dot(X_p, C_k) )  (||x||^2 dropped)
//   mode 0: accumulate per-block LDS partial sums/counts -> global atomics
//   mode 1: write int32 labels to d_out
// ---------------------------------------------------------------------------
__global__ __launch_bounds__(256) void kmeans_assign(
    const float* __restrict__ X, const __bf16* __restrict__ Cbf,
    const float* __restrict__ cc, float* __restrict__ sums,
    float* __restrict__ counts, const int* __restrict__ done,
    int* __restrict__ out_labels, int mode) {
  if (mode == 0 && *done) return;

  __shared__ float lsum[NCLUST * DIM];   // 32 KB partial sums
  __shared__ float lcount[NCLUST];
  __shared__ int   wlabels[PTS_PER_BLOCK];

  const int tid  = threadIdx.x;
  const int w    = tid >> 5;        // wave id (0..7)
  const int lane = tid & 31;
  const int half = lane >> 4;       // 0: lanes 0-15, 1: lanes 16-31
  const int lrow = lane & 15;

  if (mode == 0) {
    for (int e = tid; e < NCLUST * DIM; e += BLK) lsum[e] = 0.0f;
    if (tid < NCLUST) lcount[tid] = 0.0f;
  }

  const int base = blockIdx.x * PTS_PER_BLOCK;
  const int m0   = base + w * 16;   // this wave's 16-point strip
  const int p    = m0 + lrow;       // row this lane loads for the A fragment

  // cc values for the 4 cluster tiles this lane scores (n = t*16 + lrow)
  float ccl[4];
#pragma unroll
  for (int t = 0; t < 4; ++t) ccl[t] = cc[t * 16 + lrow];

  v8f acc[4] = {};                  // 4 N-tiles of the 16x64 dot-product block

  const float* xr = X + (size_t)p * DIM;
#pragma unroll
  for (int kb = 0; kb < DIM; kb += 32) {
    const int c0 = kb + half * 8;        // ISA 16-bit A layout: elems 0..7
    const int c1 = kb + 16 + half * 8;   //                      elems 8..15

    // A fragment: convert this point's f32 row chunks to bf16
    float4 f0 = *(const float4*)(xr + c0);
    float4 f1 = *(const float4*)(xr + c0 + 4);
    float4 f2 = *(const float4*)(xr + c1);
    float4 f3 = *(const float4*)(xr + c1 + 4);
    v16bf a;
    a[0]  = (__bf16)f0.x; a[1]  = (__bf16)f0.y; a[2]  = (__bf16)f0.z; a[3]  = (__bf16)f0.w;
    a[4]  = (__bf16)f1.x; a[5]  = (__bf16)f1.y; a[6]  = (__bf16)f1.z; a[7]  = (__bf16)f1.w;
    a[8]  = (__bf16)f2.x; a[9]  = (__bf16)f2.y; a[10] = (__bf16)f2.z; a[11] = (__bf16)f2.w;
    a[12] = (__bf16)f3.x; a[13] = (__bf16)f3.y; a[14] = (__bf16)f3.z; a[15] = (__bf16)f3.w;

#pragma unroll
    for (int t = 0; t < 4; ++t) {
      const int n = t * 16 + lrow;       // cluster column this lane supplies
      v8bf b0 = *(const v8bf*)(Cbf + (size_t)n * DIM + c0);
      v8bf b1 = *(const v8bf*)(Cbf + (size_t)n * DIM + c1);
      v16bf b;
#pragma unroll
      for (int e = 0; e < 8; ++e) { b[e] = b0[e]; b[8 + e] = b1[e]; }
      acc[t] = __builtin_amdgcn_wmma_f32_16x16x32_bf16(
          false, a, false, b, (short)0, acc[t], false, false);
    }
  }

  // Argmin over 64 clusters. C/D layout: vgpr r, half -> point m0+8*half+r,
  // lane%16 -> cluster (t*16+lrow). Butterfly min over the 16-lane half,
  // tie-break on smallest cluster index (matches jnp.argmin).
#pragma unroll
  for (int r = 0; r < 8; ++r) {
    float bestv = 3.4e38f;
    int   bestc = 0;
#pragma unroll
    for (int t = 0; t < 4; ++t) {
      float s = ccl[t] - 2.0f * acc[t][r];
      int   c = t * 16 + lrow;
      if (s < bestv || (s == bestv && c < bestc)) { bestv = s; bestc = c; }
    }
    for (int off = 1; off < 16; off <<= 1) {
      float ov = __shfl_xor(bestv, off, 32);
      int   oc = __shfl_xor(bestc, off, 32);
      if (ov < bestv || (ov == bestv && oc < bestc)) { bestv = ov; bestc = oc; }
    }
    if (lrow == 0) {
      const int lp = w * 16 + half * 8 + r;   // point index within block
      if (mode == 1) out_labels[base + lp] = bestc;
      else           wlabels[lp] = bestc;
    }
  }

  if (mode == 1) return;

  __syncthreads();

  // Segment-sum this block's 128 points into LDS partials (ds_add_f32).
  for (int idx = tid; idx < PTS_PER_BLOCK * DIM; idx += BLK) {
    const int j = idx >> 7;         // point within block
    const int d = idx & (DIM - 1);
    atomicAdd(&lsum[wlabels[j] * DIM + d], X[(size_t)(base + j) * DIM + d]);
  }
  if (tid < PTS_PER_BLOCK) atomicAdd(&lcount[wlabels[tid]], 1.0f);
  __syncthreads();

  // Flush partials: 8192+64 global f32 atomics per block.
  for (int e = tid; e < NCLUST * DIM; e += BLK) atomicAdd(&sums[e], lsum[e]);
  if (tid < NCLUST) atomicAdd(&counts[tid], lcount[tid]);
}

// ---------------------------------------------------------------------------
// update: new_C = sums / max(counts,1); move = max|C-new_C|; done if < TOL.
// Also refreshes Cbf16 + cc and zeroes sums/counts for the next iteration.
// Single block so the reduction needs no extra pass.
// ---------------------------------------------------------------------------
__global__ __launch_bounds__(256) void kmeans_update(
    float* __restrict__ C, float* __restrict__ sums,
    float* __restrict__ counts, float* __restrict__ cc,
    __bf16* __restrict__ Cbf, int* __restrict__ done) {
  if (*done) return;
  __shared__ float scnt[NCLUST];
  __shared__ float scc[NCLUST];
  __shared__ float smax[BLK];
  const int tid = threadIdx.x;
  if (tid < NCLUST) {
    scnt[tid] = fmaxf(counts[tid], 1.0f);
    scc[tid]  = 0.0f;
  }
  __syncthreads();

  float lmax = 0.0f;
  for (int e = tid; e < NCLUST * DIM; e += BLK) {
    const int k = e >> 7;
    const float nc = sums[e] / scnt[k];
    lmax = fmaxf(lmax, fabsf(C[e] - nc));
    C[e]   = nc;
    Cbf[e] = (__bf16)nc;
    sums[e] = 0.0f;
    atomicAdd(&scc[k], nc * nc);
  }
  smax[tid] = lmax;
  __syncthreads();
  for (int s = BLK / 2; s > 0; s >>= 1) {
    if (tid < s) smax[tid] = fmaxf(smax[tid], smax[tid + s]);
    __syncthreads();
  }
  if (tid < NCLUST) { cc[tid] = scc[tid]; counts[tid] = 0.0f; }
  if (tid == 0 && smax[0] < KM_TOL) *done = 1;
}

__global__ void copy_centroids(const float* __restrict__ C, float* __restrict__ out) {
  const int i = blockIdx.x * blockDim.x + threadIdx.x;
  if (i < NCLUST * DIM) out[i] = C[i];
}

// ---------------------------------------------------------------------------
extern "C" void kernel_launch(void* const* d_in, const int* in_sizes, int n_in,
                              void* d_out, int out_size, void* d_ws, size_t ws_size,
                              hipStream_t stream) {
  const float* X    = (const float*)d_in[0];   // [262144, 128] f32
  const float* prev = (const float*)d_in[1];   // [64, 128] f32

  // Workspace layout (float offsets)
  float*  ws     = (float*)d_ws;
  float*  C      = ws;                 // 8192
  float*  sums   = ws + 8192;          // 8192
  float*  counts = ws + 16384;         // 64
  float*  cc     = ws + 16448;         // 64
  int*    done   = (int*)(ws + 16512); // 1
  __bf16* Cbf    = (__bf16*)(ws + 16520); // 8192 bf16 (16B aligned)

  int*   labels = (int*)d_out;                   // first 262144 (int32 bits)
  float* outC   = (float*)d_out + N_POINTS;      // next 8192 floats

  const int nblocks = N_POINTS / PTS_PER_BLOCK;  // 2048

  kmeans_init<<<1, BLK, 0, stream>>>(prev, C, sums, counts, cc, Cbf, done);
  for (int it = 0; it < MAX_ITERS; ++it) {
    kmeans_assign<<<nblocks, BLK, 0, stream>>>(X, Cbf, cc, sums, counts, done,
                                               nullptr, 0);
    kmeans_update<<<1, BLK, 0, stream>>>(C, sums, counts, cc, Cbf, done);
  }
  // Final label pass with the converged centroids (matches reference).
  kmeans_assign<<<nblocks, BLK, 0, stream>>>(X, Cbf, cc, nullptr, nullptr, done,
                                             labels, 1);
  copy_centroids<<<(NCLUST * DIM + BLK - 1) / BLK, BLK, 0, stream>>>(C, outC);
}
